// PlasticLinearRec_81879256531092
// MI455X (gfx1250) — matrix operand order repaired and checked by reading, not compiled, not run on hardware
//
#include <hip/hip_runtime.h>
#include <hip/hip_bf16.h>

// ---------------------------------------------------------------------------
// PlasticLinearRec on MI455X (gfx1250, wave32).
//
// Roofline: hebb is 134 MB; must be read once + hebb_new written once
// => ~268 MB of HBM traffic => ~11.5 us at 23.3 TB/s. FLOPs ~270 MFLOP are
// free. Strategy: single pass over hebb (one wave per (b,j) row, hebb row
// cached in VGPRs between the reduction and the update), non-temporal
// load/store for the big stream, and the dense weight GEMM
// G = last_out @ weight^T done with v_wmma_f32_16x16x32_f16.
// ---------------------------------------------------------------------------

typedef __attribute__((ext_vector_type(16))) _Float16 v16h;
typedef __attribute__((ext_vector_type(8)))  float    v8f;
typedef __attribute__((ext_vector_type(4)))  float    v4f;   // native vec for NT builtins
typedef __attribute__((ext_vector_type(2)))  float    v2f;

#define Bsz 128
#define Fsz 512

// Kernel A: G[b,j] = sum_k last_out[b,k] * weight[j,k]  (128x512x512 GEMM)
// One wave32 per 16x16 output tile; K swept in steps of 32 via WMMA f16.
// G is written into the `out` region of d_out (kernel B reads + overwrites).
__global__ __launch_bounds__(256) void plastic_wgemm_wmma(
    const float* __restrict__ last_out,   // (B,F)
    const float* __restrict__ weight,     // (F,F) row-major: weight[j*F + k]
    float* __restrict__ G)                // (B,F)
{
    const int lane = threadIdx.x & 31;
    const int wave = threadIdx.x >> 5;
    const int tile = blockIdx.x * 8 + wave;        // 256 tiles total
    const int b0 = (tile & 7)  * 16;               // 8 row tiles (B=128)
    const int j0 = (tile >> 3) * 16;               // 32 col tiles (F=512)

    const int mn   = lane & 15;                    // M (for A) / N (for B,D)
    const int half = lane >> 4;

    v8f c = {};                                    // f32 accumulator 16x16

    // Per-lane base pointers (K-dependent offset added in loop).
    const float* arow = last_out + (size_t)(b0 + mn) * Fsz;  // A: 16x32 f16
    const float* brow = weight   + (size_t)(j0 + mn) * Fsz;  // B: 32x16 f16

    for (int k0 = 0; k0 < Fsz; k0 += 32) {
        // ---- A operand, contiguous K layout (matches B):
        //   lane<16 : elems e -> K=k0+e
        //   lane>=16: elems e -> K=k0+16+e
        const float* ap = arow + k0 + half * 16;
        v16h a;
        #pragma unroll
        for (int e = 0; e < 8; ++e) {
            a[e]     = (_Float16)ap[e];
            a[e + 8] = (_Float16)ap[8 + e];
        }
        // ---- B operand, contiguous K layout (matches A):
        //   lane<16 : elems e -> K=k0+e      (N = lane)
        //   lane>=16: elems e -> K=k0+16+e   (N = lane-16)
        const float* bp = brow + k0 + half * 16;
        v16h bm;
        #pragma unroll
        for (int e = 0; e < 16; ++e)
            bm[e] = (_Float16)bp[e];

        c = __builtin_amdgcn_wmma_f32_16x16x32_f16(
                /*neg_a=*/false, a, /*neg_b=*/false, bm,
                /*c_mod=*/(short)0, c, /*reuse_a=*/false, /*reuse_b=*/false);
    }

    // ---- D layout: VGPR r holds (M = r + 8*half, N = lane&15)
    #pragma unroll
    for (int r = 0; r < 8; ++r)
        G[(size_t)(b0 + r + 8 * half) * Fsz + (j0 + mn)] = c[r];
}

// Kernel B: fused single pass over hebb.
// Block = 256 threads = 8 waves; all 8 waves share one j (ps row in LDS),
// wave w owns (b = bt*8 + w, j). Each lane caches 16 hebb + 16 last_out
// values in VGPRs across the reduction -> update, so hebb is read once.
__global__ __launch_bounds__(256) void plastic_fused_row(
    const float* __restrict__ input,      // (B,F)
    const float* __restrict__ last_out,   // (B,F)
    const float* __restrict__ hebb,       // (B,F,F)
    const float* __restrict__ ps,         // (F,F)
    const float* __restrict__ lr_p,       // (1,)
    float* __restrict__ out,              // (B,F): holds G on entry, out on exit
    float* __restrict__ hebb_new)         // (B,F,F)
{
    __shared__ float ps_row[Fsz];

    const int tid  = threadIdx.x;
    const int wave = tid >> 5;
    const int lane = tid & 31;
    const int j    = blockIdx.x >> 4;       // 0..511
    const int bt   = blockIdx.x & 15;       // 0..15
    const int b    = bt * 8 + wave;         // 0..127

    // Stage ps[j,:] into LDS (256 threads x 2 floats).
    {
        const v2f* p2 = (const v2f*)(ps + (size_t)j * Fsz);
        ((v2f*)ps_row)[tid] = p2[tid];
    }
    __syncthreads();

    const float lr = lr_p[0];
    const float* hrow  = hebb     + ((size_t)b * Fsz + j) * Fsz;
    const float* lorow = last_out + (size_t)b * Fsz;

    v4f h[4], lo[4];
    float acc = 0.0f;
    #pragma unroll
    for (int p = 0; p < 4; ++p) {
        const int k = p * 128 + lane * 4;                  // coalesced 512B/wave
        h[p]  = __builtin_nontemporal_load((const v4f*)(hrow + k));
        lo[p] = *(const v4f*)(lorow + k);
        v4f s = *(const v4f*)(ps_row + k);
        acc += lo[p].x * (s.x * h[p].x)
             + lo[p].y * (s.y * h[p].y)
             + lo[p].z * (s.z * h[p].z)
             + lo[p].w * (s.w * h[p].w);
    }

    // wave32 butterfly reduction
    #pragma unroll
    for (int off = 16; off > 0; off >>= 1)
        acc += __shfl_xor(acc, off, 32);

    const size_t oidx = (size_t)b * Fsz + j;
    const float g = out[oidx];                 // weight-GEMM term from kernel A
    const float o = tanhf(acc + g + input[oidx]);
    if (lane == 0) out[oidx] = o;

    // hebb_new = h + lr*o*(lo - o*h) = h*(1 - lr*o*o) + (lr*o)*lo
    const float a1 = 1.0f - lr * o * o;
    const float a2 = lr * o;
    float* hnrow = hebb_new + ((size_t)b * Fsz + j) * Fsz;
    #pragma unroll
    for (int p = 0; p < 4; ++p) {
        const int k = p * 128 + lane * 4;
        v4f hn;
        hn.x = h[p].x * a1 + a2 * lo[p].x;
        hn.y = h[p].y * a1 + a2 * lo[p].y;
        hn.z = h[p].z * a1 + a2 * lo[p].z;
        hn.w = h[p].w * a1 + a2 * lo[p].w;
        __builtin_nontemporal_store(hn, (v4f*)(hnrow + k));
    }
}

extern "C" void kernel_launch(void* const* d_in, const int* in_sizes, int n_in,
                              void* d_out, int out_size, void* d_ws, size_t ws_size,
                              hipStream_t stream) {
    const float* input    = (const float*)d_in[0];   // (128,512)
    const float* last_out = (const float*)d_in[1];   // (128,512)
    const float* hebb     = (const float*)d_in[2];   // (128,512,512)
    const float* weight   = (const float*)d_in[3];   // (512,512)
    const float* ps       = (const float*)d_in[4];   // (512,512)
    const float* lr       = (const float*)d_in[5];   // (1,)

    float* out_p  = (float*)d_out;                   // first 128*512 floats
    float* hebb_p = out_p + (size_t)Bsz * Fsz;       // next 128*512*512 floats

    // Kernel A: 256 16x16 tiles, 8 waves/block -> 32 blocks. Writes G -> out_p.
    plastic_wgemm_wmma<<<32, 256, 0, stream>>>(last_out, weight, out_p);

    // Kernel B: one wave per (b,j) row: 128*512/8 = 8192 blocks of 256.
    plastic_fused_row<<<(Bsz * Fsz) / 8, 256, 0, stream>>>(
        input, last_out, hebb, ps, lr, out_p, hebb_p);
}
